// MiniKDALayer_28389733827233
// MI455X (gfx1250) — compile-verified, matrix operand order checked by reading, not run on hardware
//
#include <hip/hip_runtime.h>
#include <hip/hip_bf16.h>
#include <math.h>

#define B_   8
#define T_   4096
#define NT_  (B_*T_)
#define DIN  1024
#define DK_  64
#define DV_  64
#define DKP_ 128
#define DA_  103   // D_ALPHA
#define DAP_ 112   // padded to 16
#define DD_  39    // D_D
#define DDP_ 48    // padded to 16
#define RMS_EPS_ 1.1920928955078125e-7f
#define LN10K_OVER_DK 0.1439115683121279f   // ln(10000)/64
#define TWO_PI_ 6.283185307179586f
#define LDSA_W 1032                         // 1024 + 8 bf16 pad (bank stagger)

typedef __attribute__((ext_vector_type(16))) __bf16 v16bf;
typedef __attribute__((ext_vector_type(8)))  float  v8f;
typedef __attribute__((ext_vector_type(4)))  unsigned int u32x4;
typedef __attribute__((ext_vector_type(8)))  int i32x8;
typedef __attribute__((ext_vector_type(4)))  int i32x4;

__device__ __forceinline__ float sigmoidf_(float x){ return 1.f/(1.f+__expf(-x)); }
__device__ __forceinline__ float siluf_(float x){ return x*sigmoidf_(x); }
__device__ __forceinline__ float softplusf_(float x){ return (x>20.f)?x:log1pf(__expf(x)); }
__device__ __forceinline__ float red4_(float x){ x += __shfl_xor(x,1,32); x += __shfl_xor(x,2,32); return x; }
__device__ __forceinline__ float red8_(float x){ x += __shfl_xor(x,1,32); x += __shfl_xor(x,2,32); x += __shfl_xor(x,4,32); return x; }

// ---- WMMA fragment helpers (16x16x32 bf16, wave32 layouts per CDNA5 ISA 7.12.2) ----
__device__ __forceinline__ v16bf ldsA16x32(const __bf16* base, int ld, int row0, int k0){
  const int lane = threadIdx.x & 31;
  const __bf16* p = base + (size_t)(row0 + (lane & 15))*ld + k0 + ((lane & 16) ? 8 : 0);
  v16bf a;
#pragma unroll
  for (int g = 0; g < 8; ++g){
    int k = ((g >> 2) << 4) + ((g & 3) << 1);
    a[2*g]   = p[k];
    a[2*g+1] = p[k+1];
  }
  return a;
}

// B fragment from pre-packed bf16 weights: one lane's 16 elements contiguous (32 B) -> 2x global_load_b128.
__device__ __forceinline__ v16bf ldPackedB(const __bf16* __restrict__ P, int ntiles, int kt, int nt){
  const int lane = threadIdx.x & 31;
  return *reinterpret_cast<const v16bf*>(P + (((size_t)(kt*ntiles + nt)*32 + lane) << 4));
}

// ---- Tensor Data Mover: 1-D contiguous fp32 copy global -> LDS (D# per CDNA5 ISA ch.8) ----
// group0: [1:0]=count=1, [63:32]=lds_addr, [120:64]=global_addr, [127:126]=type=2
// group1: [17:16]=data_size=2 (4B), [79:48]=tensor_dim0, [127:112]=tile_dim0; rest 0 (<=2D, no pad/iter)
__device__ __forceinline__ void tdm_load_1d(unsigned lds_off, const float* gptr, int nelems){
  const unsigned long long ga = (unsigned long long)(uintptr_t)gptr;
  u32x4 g0;
  g0[0] = 1u;
  g0[1] = lds_off;
  g0[2] = (unsigned)ga;
  g0[3] = (unsigned)((ga >> 32) & 0x01FFFFFFu) | (2u << 30);
  i32x8 g1;
  g1[0] = 0x00020000;                       // data_size = 2 -> 4 bytes
  g1[1] = (int)((unsigned)nelems << 16);    // tensor_dim0[15:0] in bits [63:48]
  g1[2] = (int)((unsigned)nelems >> 16);    // tensor_dim0[31:16]; tensor_dim1 = 0
  g1[3] = (int)((unsigned)nelems << 16);    // tile_dim0 in bits [127:112]
  g1[4] = 0; g1[5] = 0; g1[6] = 0; g1[7] = 0;
  const i32x4 z4 = {0,0,0,0};
#if __clang_major__ >= 23
  const i32x8 z8 = {0,0,0,0,0,0,0,0};
  __builtin_amdgcn_tensor_load_to_lds(g0, g1, z4, z4, z8, 0);
#else
  __builtin_amdgcn_tensor_load_to_lds(g0, g1, z4, z4, 0);
#endif
}

// ================= Kernel P: pre-pack fp32 weight -> bf16 fragment-linear layout ======================
__global__ void __launch_bounds__(256) prepack_kernel(const float* __restrict__ W, int ldw,
                                                      int krows, int ncols, int ntiles, int total,
                                                      __bf16* __restrict__ out){
  for (int u = blockIdx.x*256 + threadIdx.x; u < total; u += gridDim.x*256){
    const int e    = u & 15;
    const int lane = (u >> 4) & 31;
    const int tile = u >> 9;
    const int nt = tile % ntiles, kt = tile / ntiles;
    const int n  = nt*16 + (lane & 15);
    const int kb = (lane & 16) ? 8 : 0;
    const int k  = kt*32 + kb + ((e >> 3) << 4) + (e & 7);
    const int kc = (k < krows) ? k : (krows - 1);       // clamp -> unconditional load
    const int nc2= (n < ncols) ? n : (ncols - 1);
    float f = W[(size_t)kc*ldw + nc2];
    f = (k < krows && n < ncols) ? f : 0.f;             // branchless zero for padding
    out[u] = (__bf16)f;
  }
}

// ================= Kernel A: fused projections  x(64x1024,bf16 in LDS) @ {q,k,v,g,up,d} ===============
__global__ void __launch_bounds__(256) proj_kernel(
    const float* __restrict__ x,
    const __bf16* __restrict__ pWq, const __bf16* __restrict__ pWk, const __bf16* __restrict__ pWv,
    const __bf16* __restrict__ pWg, const __bf16* __restrict__ pWup, const __bf16* __restrict__ pWd,
    float* __restrict__ qraw, float* __restrict__ kraw, float* __restrict__ vraw,
    float* __restrict__ gw, float* __restrict__ upw, float* __restrict__ draw)
{
  extern __shared__ __bf16 xs[];            // 64 x LDSA_W bf16 (dynamic: 132 KB of 320 KB WGP LDS)
  const int tid = threadIdx.x;
  const int token0 = blockIdx.x * 64;

  const float4* x4 = reinterpret_cast<const float4*>(x) + (size_t)token0 * (DIN/4);
  for (int u = tid; u < 64*(DIN/4); u += 256){
    int row = u >> 8;
    int c4  = u & 255;
    float4 f = x4[(size_t)row*(DIN/4) + c4];
    __bf16* dst = xs + (size_t)row*LDSA_W + c4*4;
    dst[0]=(__bf16)f.x; dst[1]=(__bf16)f.y; dst[2]=(__bf16)f.z; dst[3]=(__bf16)f.w;
  }
  __syncthreads();

  const int wave = tid >> 5;
  const int lane = tid & 31;
  for (int tile = wave; tile < 116; tile += 8){
    int t = tile;
    const __bf16* P; float* op; int ldo, ntn;
    if      (t < 16){ P=pWq;  op=qraw; ldo=64;   ntn=4; }
    else if (t < 32){ P=pWk;  op=kraw; ldo=64;   ntn=4; t-=16; }
    else if (t < 48){ P=pWv;  op=vraw; ldo=64;   ntn=4; t-=32; }
    else if (t < 76){ P=pWg;  op=gw;   ldo=DAP_; ntn=7; t-=48; }
    else if (t <104){ P=pWup; op=upw;  ldo=DAP_; ntn=7; t-=76; }
    else            { P=pWd;  op=draw; ldo=DDP_; ntn=3; t-=104;}
    const int mt = t / ntn, nt = t % ntn;
    v8f c = {};
#pragma unroll 4
    for (int kt = 0; kt < DIN/32; ++kt){
      v16bf a = ldsA16x32(xs, LDSA_W, mt*16, kt*32);
      v16bf b = ldPackedB(P, ntn, kt, nt);
      c = __builtin_amdgcn_wmma_f32_16x16x32_bf16(false, a, false, b, (short)0, c, false, false);
    }
    const int n = nt*16 + (lane & 15);
    const int rbase = token0 + mt*16 + ((lane & 16) ? 8 : 0);
#pragma unroll
    for (int r = 0; r < 8; ++r)
      op[(size_t)(rbase + r)*ldo + n] = c[r];
  }
}

// ================= Kernel B: pope(q,k), silu(v), alpha = sigmoid((silu(g)*up) @ Wdown) =================
__device__ __forceinline__ void pope_row16(const float* __restrict__ src_row, float* __restrict__ dst_row,
                                           int part, int pos, const float* __restrict__ delta_raw, bool isq){
  const float* s16 = src_row + part*16;
  float xv[16]; float ss = 0.f;
#pragma unroll
  for (int j = 0; j < 16; ++j){ xv[j] = s16[j]; ss += xv[j]*xv[j]; }
  ss = red4_(ss);
  const float rn = 1.f / fmaxf(sqrtf(ss), 1e-12f);
#pragma unroll
  for (int j = 0; j < 16; ++j){
    int i = part*16 + j;
    float mu  = softplusf_(xv[j]*rn);
    float phi = (float)pos * __expf((float)i * LN10K_OVER_DK);
    if (!isq) phi -= TWO_PI_ * sigmoidf_(delta_raw[i]);
    float sn, cs; __sincosf(phi, &sn, &cs);
    dst_row[i]       = mu*cs;
    dst_row[64 + i]  = mu*sn;
  }
}

__global__ void __launch_bounds__(256) pope_alpha_kernel(
    const float* __restrict__ qraw, const float* __restrict__ kraw, const float* __restrict__ vraw,
    const float* __restrict__ gw, const float* __restrict__ upw,
    const __bf16* __restrict__ pWdown, const float* __restrict__ delta_raw,
    float* __restrict__ qo, float* __restrict__ ko, float* __restrict__ vo, float* __restrict__ al)
{
  __shared__ __bf16 hs[64][136];
  const int tid = threadIdx.x;
  const int token0 = blockIdx.x * 64;

  {
    const int tk = tid >> 2, part = tid & 3;
    const int idx = token0 + tk;
    const int pos = idx & (T_ - 1);
    pope_row16(qraw + (size_t)idx*64, qo + (size_t)idx*128, part, pos, delta_raw, true);
    pope_row16(kraw + (size_t)idx*64, ko + (size_t)idx*128, part, pos, delta_raw, false);
  }
  for (int u = tid; u < 64*64; u += 256){
    size_t o = (size_t)token0*64 + u;
    vo[o] = siluf_(vraw[o]);
  }
  for (int u = tid; u < 64*128; u += 256){
    int row = u >> 7, cc = u & 127;
    float val = 0.f;
    if (cc < DAP_){
      size_t o = (size_t)(token0 + row)*DAP_ + cc;
      val = siluf_(gw[o]) * upw[o];
    }
    hs[row][cc] = (__bf16)val;
  }
  __syncthreads();

  const int wave = tid >> 5;
  const int lane = tid & 31;
  for (int tile = wave; tile < 32; tile += 8){
    const int mt = tile >> 3, nt = tile & 7;
    v8f c = {};
#pragma unroll
    for (int kt = 0; kt < 4; ++kt){
      v16bf a = ldsA16x32(&hs[0][0], 136, mt*16, kt*32);
      v16bf b = ldPackedB(pWdown, 8, kt, nt);
      c = __builtin_amdgcn_wmma_f32_16x16x32_bf16(false, a, false, b, (short)0, c, false, false);
    }
    const int n = nt*16 + (lane & 15);
    const int rbase = token0 + mt*16 + ((lane & 16) ? 8 : 0);
#pragma unroll
    for (int r = 0; r < 8; ++r)
      al[(size_t)(rbase + r)*DKP_ + n] = sigmoidf_(c[r]);
  }
}

// ================= Kernel C: KDA scan, per-step vectors staged by the Tensor Data Mover ================
__global__ void __launch_bounds__(512) kda_scan_kernel(
    const float* __restrict__ qo, const float* __restrict__ ko,
    const float* __restrict__ vo, const float* __restrict__ al,
    float* __restrict__ so)
{
  __shared__ float buf[2][448];                 // {q[128],k[128],a[128],v[64]}, double-buffered
  const int b = blockIdx.x;
  const int tid = threadIdx.x;
  const int e = tid >> 3;                       // value column 0..63
  const int r = tid & 7;                        // 8 lanes share one column
  const int d0 = r * 16;                        // each lane owns 16 of 128 key dims
  float S[16];
#pragma unroll
  for (int j = 0; j < 16; ++j) S[j] = 0.f;

  const size_t base0 = (size_t)b * T_;
  // wave 0 drives the TDM; other waves just consume LDS after barriers
  if (tid < 32){
    const unsigned l0 = (unsigned)(uintptr_t)(&buf[0][0]);
    tdm_load_1d(l0,        qo + base0*128, 128);
    tdm_load_1d(l0 + 512,  ko + base0*128, 128);
    tdm_load_1d(l0 + 1024, al + base0*128, 128);
    tdm_load_1d(l0 + 1536, vo + base0*64,  64);
    __builtin_amdgcn_s_wait_tensorcnt(0);
  }
  __syncthreads();

  for (int t = 0; t < T_; ++t){
    // kick off the DMA for step t+1 into the other buffer before computing step t
    if (tid < 32 && (t + 1) < T_){
      const unsigned ln = (unsigned)(uintptr_t)(&buf[(t+1) & 1][0]);
      const size_t bt = base0 + (t + 1);
      tdm_load_1d(ln,        qo + bt*128, 128);
      tdm_load_1d(ln + 512,  ko + bt*128, 128);
      tdm_load_1d(ln + 1024, al + bt*128, 128);
      tdm_load_1d(ln + 1536, vo + bt*64,  64);
    }

    const float* cur = buf[t & 1];
    const float* kk  = cur + 128;
    const float* aa  = cur + 256;
    float p = 0.f;
#pragma unroll
    for (int j = 0; j < 16; ++j) p += kk[d0+j]*aa[d0+j]*S[j];   // k . (a*S[:,e])
    p = red8_(p);
    const float delta = cur[384 + e] - p;                       // v_e - k.aS
    float o = 0.f;
#pragma unroll
    for (int j = 0; j < 16; ++j){
      float s = aa[d0+j]*S[j] + kk[d0+j]*delta;                 // S_new
      S[j] = s;
      o += cur[d0+j]*s;                                         // q . S_new
    }
    o = red8_(o);
    if (r == 0) so[(base0 + t)*64 + e] = o;

    if (tid < 32) __builtin_amdgcn_s_wait_tensorcnt(0);         // DMA for t+1 done
    __syncthreads();                                            // publish to all waves
  }
}

// ================= Kernel D: rmsnorm * rms_w, gate = sigmoid(silu(d)@Wu), out = (y*gate) @ Wout ========
__global__ void __launch_bounds__(256) out_kernel(
    const float* __restrict__ so, const float* __restrict__ draw,
    const float* __restrict__ rms_w, const __bf16* __restrict__ pWu,
    const __bf16* __restrict__ pWout, float* __restrict__ out)
{
  __shared__ float  yf[64][68];
  __shared__ __bf16 g2[64][72];
  __shared__ __bf16 yb[64][72];
  const int tid = threadIdx.x;
  const int token0 = blockIdx.x * 64;

  {
    const int tk = tid >> 2, part = tid & 3;
    const int idx = token0 + tk;
    const float* src = so + (size_t)idx*64 + part*16;
    float sv[16]; float ss = 0.f;
#pragma unroll
    for (int j = 0; j < 16; ++j){ sv[j] = src[j]; ss += sv[j]*sv[j]; }
    ss = red4_(ss);
    const float scale = rsqrtf(ss*(1.f/64.f) + RMS_EPS_);
#pragma unroll
    for (int j = 0; j < 16; ++j){
      int i = part*16 + j;
      yf[tk][i] = sv[j]*scale*rms_w[i];
    }
  }
  for (int u = tid; u < 64*64; u += 256){
    int row = u >> 6, cc = u & 63;
    float val = (cc < DDP_) ? draw[(size_t)(token0+row)*DDP_ + cc] : 0.f;
    g2[row][cc] = (__bf16)siluf_(val);
  }
  __syncthreads();

  const int wave = tid >> 5;
  const int lane = tid & 31;
  for (int tile = wave; tile < 16; tile += 8){
    const int mt = tile >> 2, nt = tile & 3;
    v8f c = {};
#pragma unroll
    for (int kt = 0; kt < 2; ++kt){
      v16bf a = ldsA16x32(&g2[0][0], 72, mt*16, kt*32);
      v16bf b = ldPackedB(pWu, 4, kt, nt);
      c = __builtin_amdgcn_wmma_f32_16x16x32_bf16(false, a, false, b, (short)0, c, false, false);
    }
    const int n  = nt*16 + (lane & 15);
    const int mb = mt*16 + ((lane & 16) ? 8 : 0);
#pragma unroll
    for (int r = 0; r < 8; ++r)
      yb[mb + r][n] = (__bf16)(yf[mb + r][n] * sigmoidf_(c[r]));
  }
  __syncthreads();

  for (int tile = wave; tile < 256; tile += 8){
    const int mt = tile & 3, nt = tile >> 2;
    v8f c = {};
#pragma unroll
    for (int kt = 0; kt < 2; ++kt){
      v16bf a = ldsA16x32(&yb[0][0], 72, mt*16, kt*32);
      v16bf b = ldPackedB(pWout, 64, kt, nt);
      c = __builtin_amdgcn_wmma_f32_16x16x32_bf16(false, a, false, b, (short)0, c, false, false);
    }
    const int n = nt*16 + (lane & 15);
    const int rbase = token0 + mt*16 + ((lane & 16) ? 8 : 0);
#pragma unroll
    for (int r = 0; r < 8; ++r)
      out[(size_t)(rbase + r)*DIN + n] = c[r];
  }
}

// =========================================== launch =====================================================
static inline void prepack(const float* W, int ldw, int krows, int ncols,
                           int ktiles, int ntiles, __bf16* out, hipStream_t stream){
  const int total = ktiles*ntiles*512;
  prepack_kernel<<<dim3((total + 255)/256), dim3(256), 0, stream>>>(W, ldw, krows, ncols, ntiles, total, out);
}

extern "C" void kernel_launch(void* const* d_in, const int* in_sizes, int n_in,
                              void* d_out, int out_size, void* d_ws, size_t ws_size,
                              hipStream_t stream){
  (void)in_sizes; (void)n_in; (void)out_size; (void)ws_size;
  const float* x     = (const float*)d_in[0];
  const float* Wq    = (const float*)d_in[1];
  const float* Wk    = (const float*)d_in[2];
  const float* Wv    = (const float*)d_in[3];
  const float* delta = (const float*)d_in[4];
  const float* Wg    = (const float*)d_in[5];
  const float* Wup   = (const float*)d_in[6];
  const float* Wdown = (const float*)d_in[7];
  const float* rms_w = (const float*)d_in[8];
  const float* Wd    = (const float*)d_in[9];
  const float* Wu    = (const float*)d_in[10];
  const float* Wout  = (const float*)d_in[11];
  float* out = (float*)d_out;
  float* ws  = (float*)d_ws;

  const size_t NT = (size_t)NT_;
  float* qraw = ws;
  float* kraw = qraw + NT*64;
  float* vraw = kraw + NT*64;
  float* gw   = vraw + NT*64;
  float* upw  = gw   + NT*DAP_;
  float* draw = upw  + NT*DAP_;
  float* qo   = draw + NT*DDP_;
  float* ko   = qo   + NT*DKP_;
  float* al   = ko   + NT*DKP_;
  float* vo   = al   + NT*DKP_;
  float* so   = vo   + NT*64;

  __bf16* pk    = (__bf16*)(so + NT*64);
  __bf16* pWq   = pk;
  __bf16* pWk   = pWq   + 32*4*512;
  __bf16* pWv   = pWk   + 32*4*512;
  __bf16* pWg   = pWv   + 32*4*512;
  __bf16* pWup  = pWg   + 32*7*512;
  __bf16* pWd   = pWup  + 32*7*512;
  __bf16* pWdn  = pWd   + 32*3*512;
  __bf16* pWu   = pWdn  + 4*8*512;
  __bf16* pWout = pWu   + 2*4*512;

  prepack(Wq,   DK_,  DIN, DK_,  32, 4,  pWq,   stream);
  prepack(Wk,   DK_,  DIN, DK_,  32, 4,  pWk,   stream);
  prepack(Wv,   DV_,  DIN, DV_,  32, 4,  pWv,   stream);
  prepack(Wg,   DA_,  DIN, DA_,  32, 7,  pWg,   stream);
  prepack(Wup,  DA_,  DIN, DA_,  32, 7,  pWup,  stream);
  prepack(Wd,   DD_,  DIN, DD_,  32, 3,  pWd,   stream);
  prepack(Wdown,DKP_, DA_, DKP_, 4,  8,  pWdn,  stream);
  prepack(Wu,   DV_,  DD_, DV_,  2,  4,  pWu,   stream);
  prepack(Wout, DIN,  DV_, DIN,  2,  64, pWout, stream);

  const dim3 blk(256);
  const dim3 grd(NT_/64);
  const size_t shA = (size_t)64 * LDSA_W * sizeof(__bf16);

  proj_kernel<<<grd, blk, shA, stream>>>(x, pWq, pWk, pWv, pWg, pWup, pWd,
                                         qraw, kraw, vraw, gw, upw, draw);
  pope_alpha_kernel<<<grd, blk, 0, stream>>>(qraw, kraw, vraw, gw, upw, pWdn, delta,
                                             qo, ko, vo, al);
  kda_scan_kernel<<<dim3(B_), dim3(512), 0, stream>>>(qo, ko, vo, al, so);
  out_kernel<<<grd, blk, 0, stream>>>(so, draw, rms_w, pWu, pWout, out);
}
